// WindowSelfAttention_80874234183880
// MI455X (gfx1250) — matrix operand order repaired and checked by reading, not compiled, not run on hardware
//
#include <hip/hip_runtime.h>

typedef __attribute__((ext_vector_type(16))) __bf16 v16bf;
typedef __attribute__((ext_vector_type(8)))  float  v8f;

union Frag { v16bf v; unsigned int u[8]; };

__device__ inline v8f wmma_bf16(v16bf a, v16bf b, v8f c) {
    return __builtin_amdgcn_wmma_f32_16x16x32_bf16(false, a, false, b, (short)0, c, false, false);
}

// ---------------- fp32 -> bf16 convert ----------------
__global__ void f2bf(const float* __restrict__ src, __bf16* __restrict__ dst, int n) {
    int i = blockIdx.x * 256 + threadIdx.x;
    if (i < n) dst[i] = (__bf16)src[i];
}

// ---------------- block reduction helper ----------------
__device__ inline void block_red2(float& a, float& b, float* redA, float* redB, int lane, int wave) {
    #pragma unroll
    for (int off = 16; off; off >>= 1) { a += __shfl_xor(a, off, 32); b += __shfl_xor(b, off, 32); }
    if (lane == 0) { redA[wave] = a; redB[wave] = b; }
    __syncthreads();
    a = 0.f; b = 0.f;
    #pragma unroll
    for (int w = 0; w < 8; ++w) { a += redA[w]; b += redB[w]; }
    __syncthreads();
}

// ---------------- pe + input LN -> bf16 x ----------------
__global__ __launch_bounds__(256) void prep_x(const float* __restrict__ hidden,
                                              const float* __restrict__ pe_w,
                                              const float* __restrict__ pe_b,
                                              const float* __restrict__ pe_g,
                                              const float* __restrict__ pe_be,
                                              const float* __restrict__ ng,
                                              const float* __restrict__ nb,
                                              __bf16* __restrict__ xbf) {
    __shared__ float red[2][8];
    const int row  = blockIdx.x;          // 0..4095 (b*2048+s)
    const int s    = row & 2047;
    const int tid  = threadIdx.x, lane = tid & 31, wave = tid >> 5;
    const float pos = ((float)s - 1024.0f) / (1024.0f + 1e-6f);

    float v[4], sum = 0.f, sq = 0.f;
    #pragma unroll
    for (int i = 0; i < 4; ++i) {
        int j = tid + 256 * i;
        float t = pos * pe_w[j] + pe_b[j];
        v[i] = t; sum += t; sq += t * t;
    }
    block_red2(sum, sq, red[0], red[1], lane, wave);
    float mu = sum * (1.0f / 1024.0f);
    float rs = rsqrtf(sq * (1.0f / 1024.0f) - mu * mu + 1e-5f);

    const float* hrow = hidden + (size_t)row * 1024;
    float t2[4], sum2 = 0.f, sq2 = 0.f;
    #pragma unroll
    for (int i = 0; i < 4; ++i) {
        int j = tid + 256 * i;
        float pe = (v[i] - mu) * rs * pe_g[j] + pe_be[j];
        float t  = hrow[j] + pe;
        t2[i] = t; sum2 += t; sq2 += t * t;
    }
    block_red2(sum2, sq2, red[0], red[1], lane, wave);
    float mu2 = sum2 * (1.0f / 1024.0f);
    float rs2 = rsqrtf(sq2 * (1.0f / 1024.0f) - mu2 * mu2 + 1e-5f);
    #pragma unroll
    for (int i = 0; i < 4; ++i) {
        int j = tid + 256 * i;
        xbf[(size_t)row * 1024 + j] = (__bf16)((t2[i] - mu2) * rs2 * ng[j] + nb[j]);
    }
}

// ---------------- LayerNorm (+optional ReLU) -> bf16 ----------------
__global__ __launch_bounds__(256) void ln_act(const float* __restrict__ z,
                                              const float* __restrict__ g,
                                              const float* __restrict__ be,
                                              __bf16* __restrict__ y, int N, int relu) {
    __shared__ float red[2][8];
    const int row = blockIdx.x;
    const int tid = threadIdx.x, lane = tid & 31, wave = tid >> 5;
    const float* zr = z + (size_t)row * N;
    float s = 0.f, s2 = 0.f;
    for (int j = tid; j < N; j += 256) { float v = zr[j]; s += v; s2 += v * v; }
    block_red2(s, s2, red[0], red[1], lane, wave);
    float mu = s / (float)N;
    float rs = rsqrtf(s2 / (float)N - mu * mu + 1e-5f);
    for (int j = tid; j < N; j += 256) {
        float v = (zr[j] - mu) * rs * g[j] + be[j];
        if (relu) v = fmaxf(v, 0.f);
        y[(size_t)row * N + j] = (__bf16)v;
    }
}

// ---------------- WMMA bf16 GEMM: C = A[M,K] @ W[K,N] + bias ----------------
// mode 0: fp32 out (row major). mode 1: bf16 out (row major).
__global__ __launch_bounds__(256) void gemm_bf16(const __bf16* __restrict__ A,
                                                 const __bf16* __restrict__ W,
                                                 const float* __restrict__ bias,
                                                 float* __restrict__ outf,
                                                 __bf16* __restrict__ outb,
                                                 int M, int N, int K, int mode) {
    __shared__ __bf16 At[128][34];   // stride 34 (17 words): conflict-free pair loads
    __shared__ __bf16 WtT[64][34];   // W tile transposed: [n][k]
    const int tid = threadIdx.x, lane = tid & 31, wave = tid >> 5;
    const int wm = wave >> 1, wn = wave & 1;
    const int m0 = blockIdx.y * 128, n0 = blockIdx.x * 64;
    const int l15 = lane & 15, hi = lane >> 4;

    v8f acc[2][2] = {};

    for (int k0 = 0; k0 < K; k0 += 32) {
        // stage A tile 128x32
        #pragma unroll
        for (int it = 0; it < 2; ++it) {
            int idx = tid + it * 256;
            int row = idx >> 2, seg = idx & 3;
            const uint4 v = *(const uint4*)(A + (size_t)(m0 + row) * K + k0 + seg * 8);
            unsigned int* dst = (unsigned int*)&At[row][seg * 8];
            dst[0] = v.x; dst[1] = v.y; dst[2] = v.z; dst[3] = v.w;
        }
        // stage W tile 32x64, transposed into WtT[n][k]
        {
            int kr = tid >> 3, seg = tid & 7;
            const uint4 v = *(const uint4*)(W + (size_t)(k0 + kr) * N + n0 + seg * 8);
            unsigned int uu[4] = { v.x, v.y, v.z, v.w };
            const __bf16* h = (const __bf16*)uu;
            #pragma unroll
            for (int j = 0; j < 8; ++j) WtT[seg * 8 + j][kr] = h[j];
        }
        __syncthreads();

        Frag af[2], bfg[2];
        #pragma unroll
        for (int mi = 0; mi < 2; ++mi) {
            int m = wm * 32 + mi * 16 + l15;
            int kh = hi * 16;
            #pragma unroll
            for (int r = 0; r < 8; ++r) {
                int kk = 2 * r + kh;
                af[mi].u[r] = *(const unsigned int*)&At[m][kk];
            }
        }
        #pragma unroll
        for (int ni = 0; ni < 2; ++ni) {
            int n = wn * 32 + ni * 16 + l15;
            int kh = hi * 16;
            #pragma unroll
            for (int r = 0; r < 8; ++r)
                bfg[ni].u[r] = *(const unsigned int*)&WtT[n][kh + 2 * r];
        }
        #pragma unroll
        for (int mi = 0; mi < 2; ++mi)
            #pragma unroll
            for (int ni = 0; ni < 2; ++ni)
                acc[mi][ni] = wmma_bf16(af[mi].v, bfg[ni].v, acc[mi][ni]);
        __syncthreads();
    }

    #pragma unroll
    for (int mi = 0; mi < 2; ++mi)
        #pragma unroll
        for (int ni = 0; ni < 2; ++ni)
            #pragma unroll
            for (int r = 0; r < 8; ++r) {
                int m = m0 + wm * 32 + mi * 16 + r + hi * 8;
                int n = n0 + wn * 32 + ni * 16 + l15;
                float v = acc[mi][ni][r] + bias[n];
                if (mode == 0) outf[(size_t)m * N + n] = v;
                else           outb[(size_t)m * N + n] = (__bf16)v;
            }
}

// ---------------- flash attention: per (b,h,q-tile of 64) ----------------
__global__ __launch_bounds__(128) void attn(const __bf16* __restrict__ Q,
                                            const __bf16* __restrict__ Kmat,
                                            const __bf16* __restrict__ V,
                                            const float* __restrict__ bias,
                                            __bf16* __restrict__ ctx) {
    __shared__ __bf16 Kt[64][66];        // [kv][hd]
    __shared__ __bf16 VtT[64][66];       // [hd][kv] (transposed)
    __shared__ __bf16 pbuf[4][16][66];   // per-wave P round-trip
    const int tid = threadIdx.x, lane = tid & 31, wave = tid >> 5;
    const int l15 = lane & 15, hi = lane >> 4;
    const int q0 = blockIdx.x * 64;
    const int bh = blockIdx.y;
    const int b = bh >> 4, h = bh & 15;
    const int qrow = q0 + wave * 16;
    const size_t Hs = 1024;
    const size_t headoff = (size_t)(b * 2048) * Hs + (size_t)h * 64;

    // Q A-fragments (K-dim = hd = 64, split into two 32-chunks)
    Frag qf[2];
    {
        int m = qrow + l15;
        int kh = hi * 16;
        #pragma unroll
        for (int c = 0; c < 2; ++c)
            #pragma unroll
            for (int r = 0; r < 8; ++r) {
                int kk = 2 * r + kh + 32 * c;
                qf[c].u[r] = *(const unsigned int*)(Q + headoff + (size_t)m * Hs + kk);
            }
    }

    float mrow[8], lrow[8];
    #pragma unroll
    for (int r = 0; r < 8; ++r) { mrow[r] = -3.0e38f; lrow[r] = 0.f; }
    v8f acc[4] = {};
    const float* biasb = bias + (size_t)h * 2049 * 2049;

    for (int kb = 0; kb < 2048; kb += 64) {
        // stage K (as-is) and V (transposed) 64x64 tiles
        for (int idx = tid; idx < 512; idx += 128) {
            int t = idx >> 3, seg = idx & 7;
            const uint4 kv = *(const uint4*)(Kmat + headoff + (size_t)(kb + t) * Hs + seg * 8);
            unsigned int* dk = (unsigned int*)&Kt[t][seg * 8];
            dk[0] = kv.x; dk[1] = kv.y; dk[2] = kv.z; dk[3] = kv.w;
            const uint4 vv = *(const uint4*)(V + headoff + (size_t)(kb + t) * Hs + seg * 8);
            unsigned int uu[4] = { vv.x, vv.y, vv.z, vv.w };
            const __bf16* hh = (const __bf16*)uu;
            #pragma unroll
            for (int j = 0; j < 8; ++j) VtT[seg * 8 + j][t] = hh[j];
        }
        __syncthreads();

        // scores S = Q K^T * 0.125 + bias
        v8f sj[4];
        #pragma unroll
        for (int j = 0; j < 4; ++j) {
            Frag kf0, kf1;
            int n = 16 * j + l15;
            int kh = hi * 16;
            #pragma unroll
            for (int r = 0; r < 8; ++r) {
                kf0.u[r] = *(const unsigned int*)&Kt[n][kh + 2 * r];
                kf1.u[r] = *(const unsigned int*)&Kt[n][kh + 2 * r + 32];
            }
            v8f z = {};
            z = wmma_bf16(qf[0].v, kf0.v, z);
            z = wmma_bf16(qf[1].v, kf1.v, z);
            #pragma unroll
            for (int r = 0; r < 8; ++r) {
                int qg = qrow + r + hi * 8;
                int kg = kb + 16 * j + l15;
                z[r] = z[r] * 0.125f + biasb[(size_t)qg * 2049 + kg];
            }
            sj[j] = z;
        }

        // online softmax (row stats live per 16-lane half, slot r = row r+8*hi)
        #pragma unroll
        for (int r = 0; r < 8; ++r) {
            float mx = fmaxf(fmaxf(sj[0][r], sj[1][r]), fmaxf(sj[2][r], sj[3][r]));
            #pragma unroll
            for (int off = 1; off < 16; off <<= 1) mx = fmaxf(mx, __shfl_xor(mx, off, 32));
            float nm = fmaxf(mrow[r], mx);
            float alpha = __expf(mrow[r] - nm);
            mrow[r] = nm;
            float rs = 0.f;
            #pragma unroll
            for (int j = 0; j < 4; ++j) { float p = __expf(sj[j][r] - nm); sj[j][r] = p; rs += p; }
            #pragma unroll
            for (int off = 1; off < 16; off <<= 1) rs += __shfl_xor(rs, off, 32);
            lrow[r] = lrow[r] * alpha + rs;
            #pragma unroll
            for (int jo = 0; jo < 4; ++jo) acc[jo][r] *= alpha;
        }

        // P -> LDS (C-layout to memory), then reload in A-layout
        #pragma unroll
        for (int j = 0; j < 4; ++j)
            #pragma unroll
            for (int r = 0; r < 8; ++r)
                pbuf[wave][r + hi * 8][16 * j + l15] = (__bf16)sj[j][r];
        __syncthreads();

        Frag pf[2];
        {
            int kh = hi * 16;
            #pragma unroll
            for (int c = 0; c < 2; ++c)
                #pragma unroll
                for (int r = 0; r < 8; ++r) {
                    int kk = 2 * r + kh + 32 * c;
                    pf[c].u[r] = *(const unsigned int*)&pbuf[wave][l15][kk];
                }
        }
        #pragma unroll
        for (int jo = 0; jo < 4; ++jo) {
            Frag vf0, vf1;
            int n = 16 * jo + l15;
            int kh = hi * 16;
            #pragma unroll
            for (int r = 0; r < 8; ++r) {
                vf0.u[r] = *(const unsigned int*)&VtT[n][kh + 2 * r];
                vf1.u[r] = *(const unsigned int*)&VtT[n][kh + 2 * r + 32];
            }
            acc[jo] = wmma_bf16(pf[0].v, vf0.v, acc[jo]);
            acc[jo] = wmma_bf16(pf[1].v, vf1.v, acc[jo]);
        }
        __syncthreads();
    }

    // normalize + store ctx (row-major [token][H], head-strided cols)
    #pragma unroll
    for (int r = 0; r < 8; ++r) {
        float inv = 1.0f / lrow[r];
        int m = qrow + r + hi * 8;
        size_t tok = (size_t)b * 2048 + m;
        #pragma unroll
        for (int jo = 0; jo < 4; ++jo)
            ctx[tok * Hs + h * 64 + 16 * jo + l15] = (__bf16)(acc[jo][r] * inv);
    }
}

extern "C" void kernel_launch(void* const* d_in, const int* in_sizes, int n_in,
                              void* d_out, int out_size, void* d_ws, size_t ws_size,
                              hipStream_t stream) {
    (void)in_sizes; (void)n_in; (void)out_size; (void)ws_size;
    const float* hidden  = (const float*)d_in[0];
    const float* wbias   = (const float*)d_in[1];
    const float* pe_w    = (const float*)d_in[2];
    const float* pe_b    = (const float*)d_in[3];
    const float* pe_g    = (const float*)d_in[4];
    const float* pe_beta = (const float*)d_in[5];
    const float* norm_g  = (const float*)d_in[6];
    const float* norm_b  = (const float*)d_in[7];
    const float* Wq = (const float*)d_in[8];  const float* bq = (const float*)d_in[9];
    const float* Wk = (const float*)d_in[10]; const float* bk = (const float*)d_in[11];
    const float* Wv = (const float*)d_in[12]; const float* bv = (const float*)d_in[13];
    const float* W1 = (const float*)d_in[14]; const float* b1 = (const float*)d_in[15];
    const float* g1 = (const float*)d_in[16]; const float* be1 = (const float*)d_in[17];
    const float* W2 = (const float*)d_in[18]; const float* b2 = (const float*)d_in[19];
    const float* g2 = (const float*)d_in[20]; const float* be2 = (const float*)d_in[21];
    const float* W3 = (const float*)d_in[22]; const float* b3 = (const float*)d_in[23];
    const float* g3 = (const float*)d_in[24]; const float* be3 = (const float*)d_in[25];
    const float* W4 = (const float*)d_in[26]; const float* b4 = (const float*)d_in[27];

    const int Hh = 1024, H2 = 2048, NT = 4096;

    char* ws = (char*)d_ws;
    size_t off = 0;
    auto alloc = [&](size_t bytes) -> char* {
        char* p = ws + off;
        off += (bytes + 255) & ~(size_t)255;
        return p;
    };
    __bf16* xbf  = (__bf16*)alloc((size_t)NT * Hh * 2);
    __bf16* wqb  = (__bf16*)alloc((size_t)Hh * Hh * 2);
    __bf16* wkb  = (__bf16*)alloc((size_t)Hh * Hh * 2);
    __bf16* wvb  = (__bf16*)alloc((size_t)Hh * Hh * 2);
    __bf16* w1b  = (__bf16*)alloc((size_t)Hh * H2 * 2);
    __bf16* w2b  = (__bf16*)alloc((size_t)H2 * H2 * 2);
    __bf16* w3b  = (__bf16*)alloc((size_t)H2 * H2 * 2);
    __bf16* w4b  = (__bf16*)alloc((size_t)H2 * Hh * 2);
    __bf16* qb   = (__bf16*)alloc((size_t)NT * Hh * 2);
    __bf16* kb_  = (__bf16*)alloc((size_t)NT * Hh * 2);
    __bf16* vb_  = (__bf16*)alloc((size_t)NT * Hh * 2);
    __bf16* ctxb = (__bf16*)alloc((size_t)NT * Hh * 2);
    float*  z    = (float*) alloc((size_t)NT * H2 * 4);
    __bf16* yA   = (__bf16*)alloc((size_t)NT * H2 * 2);
    __bf16* yB   = (__bf16*)alloc((size_t)NT * H2 * 2);

    auto conv = [&](const float* s, __bf16* d, int n) {
        f2bf<<<(n + 255) / 256, 256, 0, stream>>>(s, d, n);
    };
    conv(Wq, wqb, Hh * Hh);
    conv(Wk, wkb, Hh * Hh);
    conv(Wv, wvb, Hh * Hh);
    conv(W1, w1b, Hh * H2);
    conv(W2, w2b, H2 * H2);
    conv(W3, w3b, H2 * H2);
    conv(W4, w4b, H2 * Hh);

    prep_x<<<NT, 256, 0, stream>>>(hidden, pe_w, pe_b, pe_g, pe_beta, norm_g, norm_b, xbf);

    // QKV projections (bf16 out)
    gemm_bf16<<<dim3(Hh / 64, NT / 128), 256, 0, stream>>>(xbf, wqb, bq, nullptr, qb,  NT, Hh, Hh, 1);
    gemm_bf16<<<dim3(Hh / 64, NT / 128), 256, 0, stream>>>(xbf, wkb, bk, nullptr, kb_, NT, Hh, Hh, 1);
    gemm_bf16<<<dim3(Hh / 64, NT / 128), 256, 0, stream>>>(xbf, wvb, bv, nullptr, vb_, NT, Hh, Hh, 1);

    // fused attention with bias + online softmax
    attn<<<dim3(2048 / 64, 32), 128, 0, stream>>>(qb, kb_, vb_, wbias, ctxb);

    // MLP: GEMM -> LN+ReLU (x3) -> final GEMM to fp32 out
    gemm_bf16<<<dim3(H2 / 64, NT / 128), 256, 0, stream>>>(ctxb, w1b, b1, z, nullptr, NT, H2, Hh, 0);
    ln_act<<<NT, 256, 0, stream>>>(z, g1, be1, yA, H2, 1);
    gemm_bf16<<<dim3(H2 / 64, NT / 128), 256, 0, stream>>>(yA, w2b, b2, z, nullptr, NT, H2, H2, 0);
    ln_act<<<NT, 256, 0, stream>>>(z, g2, be2, yB, H2, 1);
    gemm_bf16<<<dim3(H2 / 64, NT / 128), 256, 0, stream>>>(yB, w3b, b3, z, nullptr, NT, H2, H2, 0);
    ln_act<<<NT, 256, 0, stream>>>(z, g3, be3, yA, H2, 1);
    gemm_bf16<<<dim3(Hh / 64, NT / 128), 256, 0, stream>>>(yA, w4b, b4, (float*)d_out, nullptr, NT, Hh, H2, 0);
}
